// QuantumLSTM_65481071399296
// MI455X (gfx1250) — matrix operand order repaired and checked by reading, not compiled, not run on hardware
//
#include <hip/hip_runtime.h>
#include <hip/hip_bf16.h>

typedef __attribute__((ext_vector_type(16))) _Float16 v16h;
typedef __attribute__((ext_vector_type(8)))  float    v8f;

#define T_STEPS 1024
#define BATCH   1024
#define INDIM   128
#define HID     16
#define DDIM    144   // INDIM + HID

// ---------------------------------------------------------------------------
// Kernel A: pack W_{f,i,g,o} [16 x 144] (row-major, fp32) into f16 WMMA
// B-operand fragments.
//   wfragX[kc][nt][lane][e], kc=0..3 (K chunks of 32 over the input part)
//   wfragH[nt][lane][e]      (recurrent part, K 0..15 real, 16..31 zero)
// B-operand layout (16-bit, 32x16): lane holds column n = lane&15,
// half-element e maps to K = (e>>3)*16 + (lane>>4)*8 + (e&7).
// ---------------------------------------------------------------------------
__global__ void qlstm_pack_w(const float* __restrict__ Wf,
                             const float* __restrict__ Wi,
                             const float* __restrict__ Wg,
                             const float* __restrict__ Wo,
                             _Float16* __restrict__ wfragX,
                             _Float16* __restrict__ wfragH) {
  for (int id = threadIdx.x; id < 4 * 4 * 32 * 16; id += blockDim.x) {
    const int e    = id & 15;
    const int lane = (id >> 4) & 31;
    const int nt   = (id >> 9) & 3;
    const int kc   = id >> 11;
    const int K    = kc * 32 + ((e >> 3) << 4) + ((lane >> 4) << 3) + (e & 7);
    const int n    = lane & 15;
    const float* W = (nt == 0) ? Wf : (nt == 1) ? Wi : (nt == 2) ? Wg : Wo;
    wfragX[id] = (_Float16)W[n * DDIM + K];
  }
  for (int id = threadIdx.x; id < 4 * 32 * 16; id += blockDim.x) {
    const int e    = id & 15;
    const int lane = (id >> 4) & 31;
    const int nt   = id >> 9;
    const int Kloc = ((e >> 3) << 4) + ((lane >> 4) << 3) + (e & 7);
    const int n    = lane & 15;
    const float* W = (nt == 0) ? Wf : (nt == 1) ? Wi : (nt == 2) ? Wg : Wo;
    wfragH[id] = (Kloc < HID) ? (_Float16)W[n * DDIM + INDIM + Kloc] : (_Float16)0.0f;
  }
}

// ---------------------------------------------------------------------------
// Kernel B: pre[t,b,:] = x[t,b,:] @ W_x^T + bias, for all 1M rows.
// One wave per 16-row M-tile, 4 N-tiles (gates f,i,g,o), K = 128 in 4 WMMA
// chunks. Result stored in raw C-fragment layout:
//   pre[((tile*4 + nt)*32 + lane)*8 .. +7]  (v8f per lane)
// A-operand layout (16-bit, 16x32): lane holds row m = lane&15,
// e<8 -> K = (lane>=16? 8:0)+e ; e>=8 -> K = 16+(lane>=16? 8:0)+(e-8).
// ---------------------------------------------------------------------------
__global__ void qlstm_pregemm(const float* __restrict__ X,
                              const _Float16* __restrict__ wfragX,
                              const float* __restrict__ bf,
                              const float* __restrict__ bi,
                              const float* __restrict__ bg,
                              const float* __restrict__ bo,
                              float* __restrict__ pre) {
  const int lane = threadIdx.x & 31;
  const int tile = blockIdx.x * 8 + (threadIdx.x >> 5);
  const int n    = lane & 15;
  const int hi8  = (lane >> 4) * 8;

  const float* xrow = X + (size_t)(tile * 16 + n) * INDIM;

  v8f c0, c1, c2, c3;
  const float b0 = bf[n], b1 = bi[n], b2 = bg[n], b3 = bo[n];
#pragma unroll
  for (int r = 0; r < 8; ++r) { c0[r] = b0; c1[r] = b1; c2[r] = b2; c3[r] = b3; }

#pragma unroll
  for (int kc = 0; kc < 4; ++kc) {
    const float4* p = (const float4*)(xrow + kc * 32 + hi8);
    const float4* q = (const float4*)(xrow + kc * 32 + 16 + hi8);
    const float4 fa = p[0], fb = p[1], ga = q[0], gb = q[1];
    const float af[16] = {fa.x, fa.y, fa.z, fa.w, fb.x, fb.y, fb.z, fb.w,
                          ga.x, ga.y, ga.z, ga.w, gb.x, gb.y, gb.z, gb.w};
    v16h a;
#pragma unroll
    for (int e = 0; e < 16; ++e) a[e] = (_Float16)af[e];

    const v16h bv0 = *(const v16h*)(wfragX + ((kc * 4 + 0) * 32 + lane) * 16);
    const v16h bv1 = *(const v16h*)(wfragX + ((kc * 4 + 1) * 32 + lane) * 16);
    const v16h bv2 = *(const v16h*)(wfragX + ((kc * 4 + 2) * 32 + lane) * 16);
    const v16h bv3 = *(const v16h*)(wfragX + ((kc * 4 + 3) * 32 + lane) * 16);
    c0 = __builtin_amdgcn_wmma_f32_16x16x32_f16(false, a, false, bv0, (short)0, c0, false, false);
    c1 = __builtin_amdgcn_wmma_f32_16x16x32_f16(false, a, false, bv1, (short)0, c1, false, false);
    c2 = __builtin_amdgcn_wmma_f32_16x16x32_f16(false, a, false, bv2, (short)0, c2, false, false);
    c3 = __builtin_amdgcn_wmma_f32_16x16x32_f16(false, a, false, bv3, (short)0, c3, false, false);
  }

  float* outp = pre + (size_t)(tile * 4) * 256 + lane * 8;
  *(v8f*)(outp + 0 * 256) = c0;
  *(v8f*)(outp + 1 * 256) = c1;
  *(v8f*)(outp + 2 * 256) = c2;
  *(v8f*)(outp + 3 * 256) = c3;
}

// ---------------------------------------------------------------------------
// Kernel C: sequential scan. 64 blocks (16 batch rows each) x 128 threads
// (4 waves, one per gate). Per step: C = pre fragment, D = h*W_h^T + C via
// one WMMA, qgate (sin + 16-lane cross-lane inclusive scan + CNOT-wrap),
// sigmoid/tanh, LDS combine of all gates -> c,h update, restage h as the
// next f16 A operand.
// ---------------------------------------------------------------------------
__global__ void qlstm_scan(const float* __restrict__ pre,
                           const _Float16* __restrict__ wfragH,
                           const float* __restrict__ thf,
                           const float* __restrict__ thi,
                           const float* __restrict__ thg,
                           const float* __restrict__ tho,
                           float* __restrict__ out) {
  __shared__ float gres[4][16][16];
  __shared__ float c_s[16][16];
  __shared__ float h_s[16][16];

  const int tid  = threadIdx.x;
  const int lane = tid & 31;
  const int nt   = tid >> 5;          // gate: 0=f 1=i 2=g 3=o
  const int n    = lane & 15;
  const int hi8  = (lane >> 4) * 8;
  const int rb   = blockIdx.x * 16;   // batch row base

  const float* thp = (nt == 0) ? thf : (nt == 1) ? thi : (nt == 2) ? thg : tho;
  const float theta = thp[n];

  const v16h bh = *(const v16h*)(wfragH + (nt * 32 + lane) * 16);
  v16h ah;
#pragma unroll
  for (int e = 0; e < 16; ++e) ah[e] = (_Float16)0.0f;   // h0 = 0; upper K zero forever

  {  // c0 = 0
    const int e0 = tid * 2;
    c_s[e0 >> 4][e0 & 15] = 0.0f;
    c_s[e0 >> 4][(e0 & 15) + 1] = 0.0f;
  }
  __syncthreads();

  const int mA = (tid * 2) >> 4;   // combine-stage element mapping (2 per thread)
  const int nA = (tid * 2) & 15;   // even, so nA and nA+1 share row mA

  for (int t = 0; t < T_STEPS; ++t) {
    const int tile = t * 64 + blockIdx.x;
    const float* cp = pre + ((size_t)(tile * 4 + nt) * 32 + lane) * 8;
    v8f acc = *(const v8f*)cp;
    if (t < T_STEPS - 1)
      __builtin_prefetch(cp + 64 * 1024, 0, 1);   // next timestep's fragment

    acc = __builtin_amdgcn_wmma_f32_16x16x32_f16(false, ah, false, bh, (short)0, acc, false, false);

    // qgate + gate nonlinearity; scan runs across the 16 lanes of each half-wave
#pragma unroll
    for (int r = 0; r < 8; ++r) {
      const float o0 = __sinf(acc[r] + theta);
      float s = o0;
#pragma unroll
      for (int d = 1; d < 16; d <<= 1) {
        const float y = __shfl_up(s, (unsigned)d, 16);
        s = (n >= d) ? s + y : s;
      }
      const float tot = __shfl(s, 15, 16);          // c[:, -1]
      const float qv  = (n == 0) ? (o0 + tot) : s;  // CNOT wrap onto column 0
      const float val = (nt == 2) ? tanhf(qv) : (1.0f / (1.0f + __expf(-qv)));
      gres[nt][r + hi8][n] = val;
    }
    __syncthreads();

    // LSTM combine: 256 elements over 128 threads
    {
      const float f0 = gres[0][mA][nA],     i0 = gres[1][mA][nA];
      const float g0 = gres[2][mA][nA],     o0g = gres[3][mA][nA];
      const float f1 = gres[0][mA][nA + 1], i1 = gres[1][mA][nA + 1];
      const float g1 = gres[2][mA][nA + 1], o1g = gres[3][mA][nA + 1];
      const float c0n = f0 * c_s[mA][nA] + i0 * g0;
      const float c1n = f1 * c_s[mA][nA + 1] + i1 * g1;
      const float h0 = o0g * tanhf(c0n);
      const float h1 = o1g * tanhf(c1n);
      c_s[mA][nA] = c0n; c_s[mA][nA + 1] = c1n;
      h_s[mA][nA] = h0;  h_s[mA][nA + 1] = h1;
      *(float2*)(out + ((size_t)t * BATCH + rb + mA) * HID + nA) = make_float2(h0, h1);
    }
    __syncthreads();

    // restage h into A-operand layout (lane = row m, e<8 -> K = hi8+e)
#pragma unroll
    for (int e = 0; e < 8; ++e) ah[e] = (_Float16)h_s[n][hi8 + e];
  }

  // final hT, cT
  const size_t TBH = (size_t)T_STEPS * BATCH * HID;
  *(float2*)(out + TBH + (size_t)(rb + mA) * HID + nA) =
      make_float2(h_s[mA][nA], h_s[mA][nA + 1]);
  *(float2*)(out + TBH + (size_t)BATCH * HID + (size_t)(rb + mA) * HID + nA) =
      make_float2(c_s[mA][nA], c_s[mA][nA + 1]);
}

// ---------------------------------------------------------------------------
extern "C" void kernel_launch(void* const* d_in, const int* in_sizes, int n_in,
                              void* d_out, int out_size, void* d_ws, size_t ws_size,
                              hipStream_t stream) {
  (void)in_sizes; (void)n_in; (void)out_size; (void)ws_size;

  const float* X   = (const float*)d_in[0];
  const float* Wf  = (const float*)d_in[1];
  const float* bf  = (const float*)d_in[2];
  const float* thf = (const float*)d_in[3];
  const float* Wi  = (const float*)d_in[4];
  const float* bi  = (const float*)d_in[5];
  const float* thi = (const float*)d_in[6];
  const float* Wg  = (const float*)d_in[7];
  const float* bg  = (const float*)d_in[8];
  const float* thg = (const float*)d_in[9];
  const float* Wo  = (const float*)d_in[10];
  const float* bo  = (const float*)d_in[11];
  const float* tho = (const float*)d_in[12];
  float* out = (float*)d_out;

  _Float16* wfragX = (_Float16*)d_ws;                       // 16 KB
  _Float16* wfragH = (_Float16*)((char*)d_ws + 16384);      // 4 KB
  float*    pre    = (float*)((char*)d_ws + 32768);         // 256 MB

  qlstm_pack_w<<<1, 256, 0, stream>>>(Wf, Wi, Wg, Wo, wfragX, wfragH);
  qlstm_pregemm<<<(T_STEPS * BATCH / 16) / 8, 256, 0, stream>>>(
      X, wfragX, bf, bi, bg, bo, pre);
  qlstm_scan<<<BATCH / 16, 128, 0, stream>>>(
      pre, wfragH, thf, thi, thg, tho, out);
}
